// GraphSAGEConv_81003083203030
// MI455X (gfx1250) — matrix operand order repaired and checked by reading, not compiled, not run on hardware
//
#include <hip/hip_runtime.h>

#define NNODES 50000
#define NEDGES 600000
#define CH 128

typedef __attribute__((ext_vector_type(16))) _Float16 v16h;
typedef __attribute__((ext_vector_type(8)))  float    v8f;
typedef __attribute__((ext_vector_type(4)))  unsigned v4u;

// TDM descriptor group types (amdgpu-toolchain 6-arg builtin form:
//   (uint32x4 g0, int32x8 g1, int32x4 g2, int32x4 g3, int32x8 gx, i32 cpol))
typedef __attribute__((ext_vector_type(4))) unsigned tdm_g0_t;
typedef __attribute__((ext_vector_type(8))) int      tdm_g1_t;
typedef __attribute__((ext_vector_type(4))) int      tdm_g23_t;

union AFrag { unsigned u[8]; v16h h; };
union BFrag { v4u q[2];      v16h h; };

// ---------------------------------------------------------------------------
// Kernel 1: zero the sum accumulator (lives in d_out) and the count array.
// ---------------------------------------------------------------------------
__global__ void sage_zero(float* __restrict__ sum, float* __restrict__ cnt) {
    const int stride = gridDim.x * blockDim.x;
    const int tid = blockIdx.x * blockDim.x + threadIdx.x;
    for (int i = tid; i < NNODES * CH; i += stride) sum[i] = 0.0f;
    for (int i = tid; i < NNODES;      i += stride) cnt[i] = 0.0f;
}

// ---------------------------------------------------------------------------
// Kernel 2: convert [Wn ; Wr] once to a f16 image wh[n][k], k in [0,256).
// 64KB total; L2-resident, feeds per-block TDM copies in the GEMM.
// ---------------------------------------------------------------------------
__global__ void sage_prep_w(const float* __restrict__ Wn,
                            const float* __restrict__ Wr,
                            _Float16* __restrict__ wh) {
    const int idx = blockIdx.x * blockDim.x + threadIdx.x;   // 0..32767
    const int n = idx >> 8, k = idx & 255;
    wh[idx] = (_Float16)((k < 128) ? Wn[n * 128 + k] : Wr[n * 128 + (k - 128)]);
}

// ---------------------------------------------------------------------------
// Kernel 3: edge scatter. One wave32 per edge: each lane moves 4 channels
// (float4 gather from x[src], 4 global f32 atomics into sum[dst]).
// x (25.6MB) is L2-resident on MI455X (192MB L2), so gathers are L2 hits.
// ---------------------------------------------------------------------------
__global__ void sage_scatter(const float* __restrict__ x,
                             const int*   __restrict__ ei,
                             float* __restrict__ sum,
                             float* __restrict__ cnt) {
    const int lane = threadIdx.x & 31;
    const int e = blockIdx.x * 8 + (threadIdx.x >> 5);
    if (e >= NEDGES) return;
    const int dst = ei[e];            // edge_index[0][e]
    const int src = ei[NEDGES + e];   // edge_index[1][e]
    const float4 v = ((const float4*)(x + (size_t)src * CH))[lane];
    float* srow = sum + (size_t)dst * CH + lane * 4;
    atomicAdd(srow + 0, v.x);
    atomicAdd(srow + 1, v.y);
    atomicAdd(srow + 2, v.z);
    atomicAdd(srow + 3, v.w);
    if (lane == 0) atomicAdd(cnt + dst, 1.0f);
}

// ---------------------------------------------------------------------------
// Kernel 4: fused GEMM  out = [sum/count | x] (Nx256) @ [Wn ; Wr]^T + bias
// Block = 256 threads (8 waves), 128 output rows per block.
// B (f16 weights, 64KB) is DMA'd into LDS by the Tensor Data Mover (wave 0),
// overlapping with the A staging done by all 8 waves on the vector pipes.
// Each wave owns 16 rows, accumulates 8 16x16 f32 tiles via
// v_wmma_f32_16x16x32_f16 (8 k-chunks x 8 n-tiles = 64 WMMA per wave).
// ---------------------------------------------------------------------------
__global__ void __launch_bounds__(256, 2)
sage_gemm(const float* __restrict__ x,
          const _Float16* __restrict__ wh,   // [128][256] f16 weight image
          const float* __restrict__ bias,
          const float* __restrict__ cnt,
          float* __restrict__ out) {
    extern __shared__ _Float16 smem[];
    _Float16* Bs = smem;              // [128][256] = 64KB (TDM destination)
    _Float16* As = smem + 128 * 256;  // [128][256] = 64KB

    const int tid  = threadIdx.x;
    const int lane = tid & 31;
    const int wv   = tid >> 5;
    const int h    = lane >> 4;   // half-wave (K-group select per ISA layout)
    const int r    = lane & 15;   // row-in-tile / col-in-tile

    // ---- wave 0: TDM copy of the 64KB f16 weight image into Bs ------------
    if (wv == 0) {
        const unsigned lds_off = (unsigned)(unsigned long long)(void*)Bs;
        const unsigned long long ga = (unsigned long long)wh;
        // D# group 0: count=1, is_restore=0, lds_addr, global_addr, type=2
        tdm_g0_t g0;
        g0[0] = 1u;
        g0[1] = lds_off;
        g0[2] = (unsigned)(ga & 0xFFFFFFFFu);
        g0[3] = (unsigned)((ga >> 32) & 0x01FFFFFFu) | (2u << 30);
        // D# group 1: data_size=3 (8B elems), 1D: tensor_dim0 = tile_dim0 =
        // 8192 elems (64KB), stride0 = 8192; dim1 = 1.
        tdm_g1_t g1;
        g1[0] = (int)(3u << 16);                    // data_size=3, no flags
        g1[1] = (int)(0x2000u << 16);               // tensor_dim0[15:0]=8192
        g1[2] = (int)(1u << 16);                    // dim0 hi=0, tensor_dim1=1
        g1[3] = (int)(0x2000u << 16);               // dim1 hi=0, tile_dim0=8192
        g1[4] = 1;                                  // tile_dim1=1, tile_dim2=0
        g1[5] = 8192;                               // tensor_dim0_stride lo32
        g1[6] = (int)(0x2000u << 16);               // stride0 hi=0, stride1 lo16
        g1[7] = 0;                                  // stride1 hi
        tdm_g23_t g2 = {1, 0, 0, 0};                // tensor_dim2=1
        tdm_g23_t g3 = {0, 0, 0, 0};
        tdm_g1_t  gx = {0, 0, 0, 0, 0, 0, 0, 0};    // reserved/extra group
        __builtin_amdgcn_tensor_load_to_lds(g0, g1, g2, g3, gx, 0);
    }

    // ---- all waves: stage their 16 rows: As[row][k] = (k<128 ? sum/cnt : x)
    const int rowbase = blockIdx.x * 128 + wv * 16;
    for (int i = 0; i < 128; ++i) {
        const int idx = i * 32 + lane;       // 0..4095 over 16 rows x 256 k
        const int lr = idx >> 8, k = idx & 255;
        int grow = rowbase + lr;
        if (grow > NNODES - 1) grow = NNODES - 1;   // clamp: keep EXEC uniform
        float val;
        if (k < 128) {
            const float c = cnt[grow];
            val = out[(size_t)grow * CH + k] * (1.0f / fmaxf(c, 1.0f));
        } else {
            val = x[(size_t)grow * CH + (k - 128)];
        }
        As[(wv * 16 + lr) * 256 + k] = (_Float16)val;
    }

    if (wv == 0) __builtin_amdgcn_s_wait_tensorcnt(0);
    __syncthreads();

    // ---- WMMA main loop ----------------------------------------------------
    v8f acc[8] = {};
    const unsigned* A32 = (const unsigned*)(As + (wv * 16 + r) * 256);
#pragma unroll
    for (int c = 0; c < 8; ++c) {
        const int kb = c * 32;
        // A fragment, 16-bit 16x32 layout (contiguous, matches B):
        //   V0..7: K = 16h + 2v + p
        AFrag a;
#pragma unroll
        for (int v = 0; v < 8; ++v) a.u[v]     = A32[(kb + 16 * h + 2 * v) >> 1];
#pragma unroll
        for (int t = 0; t < 8; ++t) {
            // B fragment, 32x16: lane holds N = r, 16 contiguous K (16h offset)
            const int n = t * 16 + r;
            BFrag b;
            const v4u* B128 = (const v4u*)(Bs + n * 256 + kb + 16 * h);
            b.q[0] = B128[0];
            b.q[1] = B128[1];
            acc[t] = __builtin_amdgcn_wmma_f32_16x16x32_f16(
                false, a.h, false, b.h, (short)0, acc[t], false, false);
        }
    }

    // ---- epilogue: C/D layout VGPR j -> row = j + 8h, col = r (+ bias) -----
#pragma unroll
    for (int t = 0; t < 8; ++t) {
        const int col = t * 16 + r;
        const float bv = bias[col];
#pragma unroll
        for (int j = 0; j < 8; ++j) {
            const int row = rowbase + j + 8 * h;
            if (row < NNODES) out[(size_t)row * CH + col] = acc[t][j] + bv;
        }
    }
}

// ---------------------------------------------------------------------------
extern "C" void kernel_launch(void* const* d_in, const int* in_sizes, int n_in,
                              void* d_out, int out_size, void* d_ws, size_t ws_size,
                              hipStream_t stream) {
    const float* x    = (const float*)d_in[0];
    const int*   ei   = (const int*)  d_in[1];
    const float* Wn   = (const float*)d_in[2];
    const float* Wr   = (const float*)d_in[3];
    const float* bias = (const float*)d_in[4];
    float* out = (float*)d_out;       // doubles as the segment-sum accumulator
    float*    cnt = (float*)d_ws;                               // 200KB
    _Float16* wh  = (_Float16*)((char*)d_ws + 256 * 1024);      // 64KB f16 weights

    sage_zero<<<2048, 256, 0, stream>>>(out, cnt);
    sage_prep_w<<<128, 256, 0, stream>>>(Wn, Wr, wh);
    sage_scatter<<<(NEDGES + 7) / 8, 256, 0, stream>>>(x, ei, out, cnt);
    sage_gemm<<<(NNODES + 127) / 128, 256, 2 * 128 * 256 * sizeof(_Float16), stream>>>(
        x, wh, bias, cnt, out);
}
